// SkipGRU_42958262894663
// MI455X (gfx1250) — compile-verified, matrix operand order checked by reading, not run on hardware
//
#include <hip/hip_runtime.h>

// ---------------------------------------------------------------------------
// Skip-GRU for MI455X (gfx1250, wave32, WMMA bf16)
//   T=384, B=48, C=H=256, 3H=768, L=2
// Plan:
//   1) convert x / w_ih / w_hh to bf16 (workspace)
//   2) per layer: bulk GEMM  G = X_bf16 @ W_ih^T + b_ih   (WMMA 16x16x32 bf16)
//   3) per layer: persistent recurrent scan kernel, 4 WGs x 384 threads,
//      W_hh slice resident in LDS, grid barriers for h / skip-state exchange
//   LDS staging uses GLOBAL_LOAD_ASYNC_TO_LDS_B128 (ASYNCcnt) when available.
// ---------------------------------------------------------------------------

typedef __bf16 bf16_t;
typedef bf16_t v16bf __attribute__((ext_vector_type(16)));
typedef float  v8f   __attribute__((ext_vector_type(8)));
typedef unsigned int u32x4 __attribute__((ext_vector_type(4)));

#define T_STEPS 384
#define BATCH   48
#define CDIM    256
#define HDIM    256
#define G3H     768
#define NWG     4
#define SCAN_THREADS 384

union V16 { v16bf v; u32x4 q[2]; };

#if defined(__has_builtin)
#if __has_builtin(__builtin_amdgcn_global_load_async_to_lds_b128)
#define HAVE_ASYNC_B128 1
#endif
#endif

#ifdef HAVE_ASYNC_B128
// Builtin signature (from hipcc diagnostic): params are pointers to
// int __attribute__((vector_size(16))) in AS1 (global) / AS3 (LDS).
typedef int vi4 __attribute__((vector_size(16)));
typedef __attribute__((address_space(1))) vi4 gvi4;
typedef __attribute__((address_space(3))) vi4 lvi4;
#endif

// 16-byte global -> LDS copy; async DMA path on gfx1250, sync fallback.
__device__ __forceinline__ void copy_b128_to_lds(const void* gsrc, void* lds) {
#ifdef HAVE_ASYNC_B128
    __builtin_amdgcn_global_load_async_to_lds_b128((gvi4*)gsrc, (lvi4*)lds,
                                                   0, 0);
#else
    *(u32x4*)lds = *(const u32x4*)gsrc;
#endif
}
__device__ __forceinline__ void copy_lds_wait() {
#ifdef HAVE_ASYNC_B128
#if __has_builtin(__builtin_amdgcn_s_wait_asynccnt)
    __builtin_amdgcn_s_wait_asynccnt(0);
#else
    asm volatile("s_wait_asynccnt 0x0" ::: "memory");
#endif
#endif
}

__device__ __forceinline__ unsigned short f2bfbits(float f) {
    unsigned u = __float_as_uint(f);
    unsigned r = ((u >> 16) & 1u) + 0x7FFFu;   // round-to-nearest-even
    return (unsigned short)((u + r) >> 16);
}
__device__ __forceinline__ float sigmoidf_(float x) {
    return 1.0f / (1.0f + __expf(-x));
}
__device__ __forceinline__ v8f wmma_bf16(v16bf a, v16bf b, v8f c) {
    return __builtin_amdgcn_wmma_f32_16x16x32_bf16(
        /*neg_a=*/false, a, /*neg_b=*/false, b,
        /*c_mod=*/(short)0, c, /*reuse_a=*/false, /*reuse_b=*/false);
}

// ---------------------------------------------------------------------------
__global__ void f32_to_bf16_kernel(const float* __restrict__ src,
                                   unsigned short* __restrict__ dst, int n) {
    for (int i = blockIdx.x * blockDim.x + threadIdx.x; i < n;
         i += gridDim.x * blockDim.x)
        dst[i] = f2bfbits(src[i]);
}

// ---------------------------------------------------------------------------
__global__ void scan_init_kernel(const float* __restrict__ h0,   // (256)
                                 unsigned short* __restrict__ hbf,
                                 float* __restrict__ dotb,
                                 unsigned* __restrict__ bar) {
    int tid = blockIdx.x * blockDim.x + threadIdx.x;
    if (tid < BATCH * HDIM) hbf[tid] = f2bfbits(h0[tid & (HDIM - 1)]);
    if (tid < 2 * BATCH)    dotb[tid] = 0.0f;
    if (tid == 0)           *bar = 0u;
}

// ---------------------------------------------------------------------------
// Bulk input GEMM:  G[m][n] = sum_k X[m][k] * W[n][k] + bias[n]
// M = T*B = 18432, K = 256, N = 768.  Block = 8 waves, 128(M) x 64(N) tile.
__global__ __launch_bounds__(256) void gi_gemm_kernel(
    const unsigned short* __restrict__ Xbf,   // (M,256) bf16
    const unsigned short* __restrict__ Wbf,   // (768,256) bf16 row-major
    const float* __restrict__ bias,           // (768)
    float* __restrict__ G)                    // (M,768)
{
    __shared__ unsigned short Bs[64 * 256];   // 32 KB weight slice
    const int tid  = threadIdx.x;
    const int lane = tid & 31;
    const int wv   = tid >> 5;
    const int hi   = lane >> 4, lo = lane & 15;
    const int mtile = blockIdx.x * 8 + wv;    // 0..1151
    const int nblk  = blockIdx.y;             // 0..11

    // stage contiguous 64x256 bf16 weight block (32 KB) via async DMA to LDS
    {
        const u32x4* src = (const u32x4*)(Wbf + (size_t)nblk * 64 * 256);
        u32x4* dst = (u32x4*)Bs;
#pragma unroll
        for (int i = 0; i < 8; ++i)
            copy_b128_to_lds(&src[tid + i * 256], &dst[tid + i * 256]);
        copy_lds_wait();
    }
    __syncthreads();

    v8f zero{};
    v8f acc[4];
#pragma unroll
    for (int ni = 0; ni < 4; ++ni) acc[ni] = zero;

    const unsigned short* arow = Xbf + (size_t)(mtile * 16 + lo) * 256;
#pragma unroll
    for (int kk = 0; kk < 8; ++kk) {
        const int kb = kk * 32 + hi * 8;
        V16 a;
        a.q[0] = *(const u32x4*)(arow + kb);
        a.q[1] = *(const u32x4*)(arow + kb + 16);
#pragma unroll
        for (int ni = 0; ni < 4; ++ni) {
            const unsigned short* brow = Bs + (ni * 16 + lo) * 256;
            V16 b;
            b.q[0] = *(const u32x4*)(brow + kb);
            b.q[1] = *(const u32x4*)(brow + kb + 16);
            acc[ni] = wmma_bf16(a.v, b.v, acc[ni]);
        }
    }
#pragma unroll
    for (int ni = 0; ni < 4; ++ni) {
        const int n = nblk * 64 + ni * 16 + lo;
        const float bv = bias[n];
#pragma unroll
        for (int r = 0; r < 8; ++r) {
            const int m = mtile * 16 + hi * 8 + r;
            G[(size_t)m * G3H + n] = acc[ni][r] + bv;
        }
    }
}

// ---------------------------------------------------------------------------
__device__ __forceinline__ void grid_barrier(unsigned* bar, unsigned target) {
    __threadfence();
    __syncthreads();
    if (threadIdx.x == 0) {
        atomicAdd(bar, 1u);
        volatile unsigned* vb = (volatile unsigned*)bar;
        while (*vb < target) { __builtin_amdgcn_s_sleep(2); }
    }
    __syncthreads();
    __threadfence();
}

// ---------------------------------------------------------------------------
// Persistent recurrent scan. 4 workgroups, each owns 64 hidden units
// (gh columns {j0..j0+63, 256+j0.., 512+j0..} = 192 cols). W_hh slice lives
// in LDS for the whole scan; h staged to LDS (bf16) each step via async DMA.
__global__ __launch_bounds__(SCAN_THREADS) void scan_kernel(
    int layer,
    const float* __restrict__ G,               // (T,B,768): x@Wih^T + b_ih
    const unsigned short* __restrict__ Whbf,   // (768,256) bf16, this layer
    const float* __restrict__ bhh,             // (768)
    const float* __restrict__ lw,              // (256)
    const float* __restrict__ lb,              // (1)
    const float* __restrict__ h0,              // (256) initial hidden
    unsigned short* __restrict__ hbf,          // (48,256) global current h
    float* __restrict__ dotbuf,                // (2,48) double-buffered dots
    unsigned* __restrict__ bar,
    unsigned short* __restrict__ ys_bf,        // (T,B,256) bf16 (layer 0 out)
    float* __restrict__ outp,                  // (T,B,256) f32  (layer 1 out)
    float* __restrict__ hs_out,                // (L,B,256)
    float* __restrict__ tu_out)                // (B, L*T)
{
    extern __shared__ char smem[];
    unsigned short* whL  = (unsigned short*)smem;                    // 96 KB
    unsigned short* hb   = (unsigned short*)(smem + 98304);          // 24 KB
    float* ghL  = (float*)(smem + 98304 + 24576);                    // 36 KB
    float* hown = (float*)(smem + 98304 + 24576 + 36864);            // 12 KB
    float* uL   = (float*)(smem + 172032);                           // 48
    float* dL   = uL + 48;
    float* buL  = uL + 96;
    float* pdot = uL + 144;
    int*   skL  = (int*)(uL + 192);

    const int tid  = threadIdx.x;
    const int lane = tid & 31;
    const int wv   = tid >> 5;                 // 0..11
    const int hi   = lane >> 4, lo = lane & 15;
    const int j0   = blockIdx.x * 64;
    const float lbv = lb[0];

    // --- load resident W_hh slice: 192 rows x 256 bf16 (3 gates x 64 units)
    for (int i = tid; i < 192 * 32; i += SCAN_THREADS) {
        const int n = i >> 5, s = i & 31;      // row, 16B-chunk within row
        const int gate = n >> 6, jl = n & 63;
        const u32x4* src =
            (const u32x4*)(Whbf + (size_t)(gate * 256 + j0 + jl) * 256);
        copy_b128_to_lds(&src[s], &((u32x4*)(whL + n * 256))[s]);
    }
    copy_lds_wait();
    // --- private f32 copy of our slice of h, per-batch skip state
    for (int i = tid; i < BATCH * 64; i += SCAN_THREADS)
        hown[i] = h0[j0 + (i & 63)];
    if (tid < BATCH) { uL[tid] = 1.0f; dL[tid] = 0.0f; skL[tid] = 0; }
    unsigned bcount = 0;
    __syncthreads();

    for (int t = 0; t < T_STEPS; ++t) {
        // ---- phase 0: async-stage full h (bf16) into LDS; bu = round(u) ---
        {
            const u32x4* src = (const u32x4*)hbf;   // 1536 x 16B
            u32x4* dst = (u32x4*)hb;
#pragma unroll
            for (int i = 0; i < 4; ++i)
                copy_b128_to_lds(&src[tid + i * SCAN_THREADS],
                                 &dst[tid + i * SCAN_THREADS]);
        }
        if (tid < BATCH) {
            const float bu = rintf(uL[tid]);        // round half-to-even
            buL[tid]  = bu;
            pdot[tid] = 0.0f;
            if (blockIdx.x == 0)
                tu_out[tid * (2 * T_STEPS) + layer * T_STEPS + t] = bu;
        }
        if (t + 1 < T_STEPS) {
            const float* gnx = G + (size_t)(t + 1) * BATCH * G3H;
            __builtin_prefetch(gnx + tid * 96, 0, 0);   // warm next step's gi
        }
        copy_lds_wait();                           // DMA overlapped with above
        __syncthreads();

        // ---- phase 1: gh = h @ Wh_slice^T  (36 tiles over 12 waves) ----
#pragma unroll
        for (int q = 0; q < 3; ++q) {
            const int tIdx = wv * 3 + q;            // 0..35
            const int mi = tIdx % 3, ni = tIdx / 3;
            v8f c{};
            const unsigned short* ar = hb  + (mi * 16 + lo) * 256;
            const unsigned short* br = whL + (ni * 16 + lo) * 256;
#pragma unroll
            for (int kk = 0; kk < 8; ++kk) {
                const int kb = kk * 32 + hi * 8;
                V16 a, b;
                a.q[0] = *(const u32x4*)(ar + kb);
                a.q[1] = *(const u32x4*)(ar + kb + 16);
                b.q[0] = *(const u32x4*)(br + kb);
                b.q[1] = *(const u32x4*)(br + kb + 16);
                c = wmma_bf16(a.v, b.v, c);
            }
            const int nl = ni * 16 + lo;
#pragma unroll
            for (int r = 0; r < 8; ++r) {
                const int m = mi * 16 + hi * 8 + r; // C/D layout: M = r + 8*hi
                ghL[m * 192 + nl] = c[r];
            }
        }
        __syncthreads();

        // ---- phase 2: GRU nonlinearity + skip update for our 64 units ----
        {
            const int b = tid >> 3;                 // 0..47
            const int jbase = (tid & 7) * 8;        // 8 units per thread
            const float bu = buL[b];
            const int   sk = skL[b];
            const float* gi = G + ((size_t)t * BATCH + b) * G3H;
            float pd = 0.0f;
#pragma unroll
            for (int q = 0; q < 8; ++q) {
                const int jl = jbase + q;
                const int j  = j0 + jl;
                const float gr = ghL[b * 192 + jl]        + bhh[j];
                const float gz = ghL[b * 192 + 64 + jl]   + bhh[256 + j];
                const float gn = ghL[b * 192 + 128 + jl]  + bhh[512 + j];
                const float rr = sigmoidf_(gi[j] + gr);
                const float zz = sigmoidf_(gi[256 + j] + gz);
                const float nn = tanhf(gi[512 + j] + rr * gn);
                const float hold = hown[b * 64 + jl];
                const float cand = (1.0f - zz) * nn + zz * hold;
                const float hns  = cand * bu;
                pd += hns * lw[j];
                const float hnew = sk ? hold * (1.0f - bu) : hns;
                hown[b * 64 + jl] = hnew;
                hbf[b * 256 + j]  = f2bfbits(hnew);
                if (layer == 0)
                    ys_bf[((size_t)t * BATCH + b) * HDIM + j] = f2bfbits(hnew);
                else
                    outp[((size_t)t * BATCH + b) * HDIM + j] = hnew;
                if (t == T_STEPS - 1)
                    hs_out[(size_t)layer * BATCH * HDIM + b * HDIM + j] = hnew;
            }
            atomicAdd(&pdot[b], pd);                // LDS ds_add_f32
        }
        __syncthreads();
        if (tid < BATCH)
            atomicAdd(&dotbuf[(t & 1) * BATCH + tid], pdot[tid]);

        // ---- barrier 1: all partial dots + h-slices globally visible ----
        ++bcount; grid_barrier(bar, NWG * bcount);

        // ---- replicated per-batch skip-state update ----
        if (tid < BATCH) {
            volatile float* vd = (volatile float*)dotbuf;
            const float dsum = vd[(t & 1) * BATCH + tid];
            const float dns  = sigmoidf_(dsum + lbv);
            const float bu = buL[tid];
            const int   sk = skL[tid];
            const float uo = uL[tid], dd = dL[tid];
            dL[tid] = sk ? dd : dns;
            const float nu =
                sk ? fminf(fmaxf(uo + dd, 0.0f), 1.0f) * (1.0f - bu)
                   : dns * bu;
            uL[tid]  = nu;
            skL[tid] = ((ceilf(0.5f / nu) - 1.0f) > 0.0f) ? 1 : 0;
        }

        // ---- barrier 2: everyone consumed dotbuf; WG0 recycles it ----
        ++bcount; grid_barrier(bar, NWG * bcount);
        if (blockIdx.x == 0 && tid < BATCH)
            dotbuf[(t & 1) * BATCH + tid] = 0.0f;
    }
}

// ---------------------------------------------------------------------------
extern "C" void kernel_launch(void* const* d_in, const int* in_sizes, int n_in,
                              void* d_out, int out_size, void* d_ws,
                              size_t ws_size, hipStream_t stream) {
    const float* x   = (const float*)d_in[0];
    const float* hid = (const float*)d_in[1];
    const float* wih = (const float*)d_in[2];
    const float* whh = (const float*)d_in[3];
    const float* bih = (const float*)d_in[4];
    const float* bhh = (const float*)d_in[5];
    const float* lw  = (const float*)d_in[6];
    const float* lb  = (const float*)d_in[7];

    char* ws = (char*)d_ws;
    unsigned short* xbf  = (unsigned short*)(ws + 0);          // 9,437,184 B
    unsigned short* ysbf = (unsigned short*)(ws + 9437184);    // 9,437,184 B
    unsigned short* wibf = (unsigned short*)(ws + 18874368);   //   786,432 B
    unsigned short* whbf = (unsigned short*)(ws + 19660800);   //   786,432 B
    float* G             = (float*)(ws + 20447232);            // 56,623,104 B
    unsigned short* hbf  = (unsigned short*)(ws + 77070336);   //    24,576 B
    float* dotb          = (float*)(ws + 77094912);            //       384 B
    unsigned* bar        = (unsigned*)(ws + 77095424);

    float* outp = (float*)d_out;
    float* hs   = outp + (size_t)T_STEPS * BATCH * HDIM;
    float* tu   = hs + 2 * (size_t)BATCH * HDIM;

    (void)hipFuncSetAttribute((const void*)scan_kernel,
                              hipFuncAttributeMaxDynamicSharedMemorySize,
                              176128);

    f32_to_bf16_kernel<<<1024, 256, 0, stream>>>(x, xbf, T_STEPS * BATCH * CDIM);
    f32_to_bf16_kernel<<<512, 256, 0, stream>>>(wih, wibf, 2 * G3H * CDIM);
    f32_to_bf16_kernel<<<512, 256, 0, stream>>>(whh, whbf, 2 * G3H * HDIM);

    for (int layer = 0; layer < 2; ++layer) {
        scan_init_kernel<<<48, 256, 0, stream>>>(hid + layer * HDIM, hbf,
                                                 dotb, bar);
        gi_gemm_kernel<<<dim3(144, 12), 256, 0, stream>>>(
            layer == 0 ? xbf : ysbf, wibf + (size_t)layer * G3H * CDIM,
            bih + layer * G3H, G);
        scan_kernel<<<NWG, SCAN_THREADS, 172992, stream>>>(
            layer, G, whbf + (size_t)layer * G3H * HDIM, bhh + layer * G3H,
            lw + layer * HDIM, lb + layer, hid + layer * HDIM, hbf, dotb, bar,
            ysbf, outp, hs, tu);
    }
}
